// PixtralViT_12481174962902
// MI455X (gfx1250) — matrix operand order repaired and checked by reading, not compiled, not run on hardware
//
#include <hip/hip_runtime.h>
#include <hip/hip_bf16.h>
#include <math.h>

// ---------------------------------------------------------------------------
// PixtralViT forward for MI455X (gfx1250): bf16 WMMA GEMMs + fp32 accum.
// B=2, S=1024, D=1024, FF=4096, L=8, H=16, dh=64, patch grid 32x32 (Hp=Wp=32).
// Weights are converted fp32->bf16 once per layer (bandwidth-trivial) so the
// GEMM inner loop is pure loads + v_wmma (no VALU conversion on the hot path).
// ---------------------------------------------------------------------------

typedef __attribute__((ext_vector_type(16))) __bf16         v16bf;
typedef __attribute__((ext_vector_type(8)))  float          v8f;
typedef __attribute__((ext_vector_type(8)))  unsigned short v8us;

union Frag16 { v16bf v; v8us h[2]; unsigned short s[16]; };

__device__ __forceinline__ unsigned short f2bfu(float f) {
  union { float f; unsigned u; } c; c.f = f;
  unsigned r = c.u + 0x7FFFu + ((c.u >> 16) & 1u);   // round-to-nearest-even
  return (unsigned short)(r >> 16);
}
__device__ __forceinline__ float bf2f(unsigned short s) {
  union { unsigned u; float f; } c; c.u = ((unsigned)s) << 16;
  return c.f;
}

// ---------------------------------------------------------------------------
// fp32 -> bf16 bulk conversion (weights), 4 elements / thread.
// ---------------------------------------------------------------------------
__global__ __launch_bounds__(256)
void cvt_kernel(const float* __restrict__ src, unsigned short* __restrict__ dst) {
  const int i = blockIdx.x * blockDim.x + threadIdx.x;
  const float4 v = ((const float4*)src)[i];
  ushort4 u;
  u.x = f2bfu(v.x); u.y = f2bfu(v.y); u.z = f2bfu(v.z); u.w = f2bfu(v.w);
  ((ushort4*)dst)[i] = u;
}

// ---------------------------------------------------------------------------
// 2D RoPE: x [B,S,D] fp32 -> h [B,S,D] fp32.  One thread per quad (D/4=256).
// ---------------------------------------------------------------------------
__global__ __launch_bounds__(256)
void rope_kernel(const float* __restrict__ x, float* __restrict__ h) {
  int idx  = blockIdx.x * blockDim.x + threadIdx.x;   // 0 .. B*S*256-1
  int q    = idx & 255;                                // quad index (d4)
  int bs   = idx >> 8;
  int s    = bs & 1023;
  int prow = s >> 5;                                   // Hp = 32
  int pcol = s & 31;                                   // Wp = 32
  float f  = __powf(10000.0f, -(float)q * (1.0f / 256.0f));
  float ah = (float)prow * f, aw = (float)pcol * f;
  float chv = __cosf(ah), shv = __sinf(ah);
  float cwv = __cosf(aw), swv = __sinf(aw);
  float4 xv = ((const float4*)x)[idx];
  float4 hv;
  hv.x = xv.x * chv - xv.y * shv;
  hv.y = xv.x * shv + xv.y * chv;
  hv.z = xv.z * cwv - xv.w * swv;
  hv.w = xv.z * swv + xv.w * cwv;
  ((float4*)h)[idx] = hv;
}

// ---------------------------------------------------------------------------
// LayerNorm over D=1024: one block (256 thr) per token. OUTBF=1 -> bf16 out.
// ---------------------------------------------------------------------------
template <int OUTBF>
__global__ __launch_bounds__(256)
void ln_kernel(const float* __restrict__ x, const float* __restrict__ g,
               const float* __restrict__ bvec,
               unsigned short* __restrict__ ob, float* __restrict__ of) {
  __shared__ float s1[256], s2[256];
  const int tok = blockIdx.x, t = threadIdx.x;
  const float4 v = ((const float4*)(x + (size_t)tok * 1024))[t];
  float sm = v.x + v.y + v.z + v.w;
  float sq = v.x * v.x + v.y * v.y + v.z * v.z + v.w * v.w;
  s1[t] = sm; s2[t] = sq;
  __syncthreads();
  for (int o = 128; o > 0; o >>= 1) {
    if (t < o) { s1[t] += s1[t + o]; s2[t] += s2[t + o]; }
    __syncthreads();
  }
  const float mean = s1[0] * (1.0f / 1024.0f);
  const float var  = s2[0] * (1.0f / 1024.0f) - mean * mean;
  const float inv  = rsqrtf(var + 1e-5f);
  const float4 gg = ((const float4*)g)[t];
  const float4 bb = ((const float4*)bvec)[t];
  float o0 = (v.x - mean) * inv * gg.x + bb.x;
  float o1 = (v.y - mean) * inv * gg.y + bb.y;
  float o2 = (v.z - mean) * inv * gg.z + bb.z;
  float o3 = (v.w - mean) * inv * gg.w + bb.w;
  if (OUTBF) {
    ushort4 u; u.x = f2bfu(o0); u.y = f2bfu(o1); u.z = f2bfu(o2); u.w = f2bfu(o3);
    ((ushort4*)(ob + (size_t)tok * 1024))[t] = u;
  } else {
    float4 ov; ov.x = o0; ov.y = o1; ov.z = o2; ov.w = o3;
    ((float4*)(of + (size_t)tok * 1024))[t] = ov;
  }
}

// ---------------------------------------------------------------------------
// WMMA GEMM: C[M,N] = Abf[M,K](bf16) @ Wbf[N,K]^T(bf16) + bias, fp32 accum.
// Block = 256 thr = 8 waves arranged 4(M) x 2(N); block tile 256x128.
// Wave tile 64x64 = 4x4 v_wmma_f32_16x16x32_bf16 per 32-deep k-step.
// MODE 0: store fp32   MODE 1: store bf16   MODE 2: Cf += result (residual)
// ---------------------------------------------------------------------------
template <int MODE>
__global__ __launch_bounds__(256)
void gemm_bf16(const unsigned short* __restrict__ Abf,
               const unsigned short* __restrict__ Wbf,
               const float* __restrict__ bias,
               float* __restrict__ Cf, unsigned short* __restrict__ Cb,
               int M, int N, int K) {
  const int lane = threadIdx.x & 31;
  const int wid  = threadIdx.x >> 5;
  const int r    = lane & 15;     // row/col within 16-subtile
  const int hk   = lane >> 4;     // half-wave selector
  const int rowBase = blockIdx.y * 256 + (wid >> 1) * 64;
  const int colBase = blockIdx.x * 128 + (wid & 1) * 64;
  (void)M;

  v8f acc[4][4];
#pragma unroll
  for (int i = 0; i < 4; i++)
#pragma unroll
    for (int j = 0; j < 4; j++)
      acc[i][j] = (v8f){0.f, 0.f, 0.f, 0.f, 0.f, 0.f, 0.f, 0.f};

  // A-fragment rows: lanes<16 K[0..15],
  // lanes>=16 K[16..31] (matches B's contiguous k order).
  const unsigned short* arow[4];
#pragma unroll
  for (int i = 0; i < 4; i++) arow[i] = Abf + (size_t)(rowBase + i * 16 + r) * K;
  // B-fragment: lane r = output column; half-wave covers K 0..15 / 16..31.
  const unsigned short* wrow[4];
#pragma unroll
  for (int j = 0; j < 4; j++) wrow[j] = Wbf + (size_t)(colBase + j * 16 + r) * K;

  for (int k0 = 0; k0 < K; k0 += 32) {
    if (k0 + 512 < K) {
      __builtin_prefetch(wrow[0] + k0 + 512, 0, 1);  // global_prefetch_b8
      __builtin_prefetch(arow[0] + k0 + 512, 0, 1);
    }
    Frag16 fa[4];
#pragma unroll
    for (int i = 0; i < 4; i++) {
      const v8us* p = (const v8us*)(arow[i] + k0 + hk * 16);
      fa[i].h[0] = p[0];           // K lo 8
      fa[i].h[1] = p[1];           // next 8 contiguous elements
    }
#pragma unroll
    for (int j = 0; j < 4; j++) {
      Frag16 fb;
      const v8us* q = (const v8us*)(wrow[j] + k0 + hk * 16);
      fb.h[0] = q[0];              // 16 contiguous bf16 = 32B
      fb.h[1] = q[1];
#pragma unroll
      for (int i = 0; i < 4; i++) {
        acc[i][j] = __builtin_amdgcn_wmma_f32_16x16x32_bf16(
            false, fa[i].v, false, fb.v, (short)0, acc[i][j], false, false);
      }
    }
  }

  // Epilogue. C/D layout: VGPR v -> M = v + hk*8 (per 16-subtile), col = r.
#pragma unroll
  for (int i = 0; i < 4; i++) {
#pragma unroll
    for (int j = 0; j < 4; j++) {
      const int col = colBase + j * 16 + r;
      const float bj = bias[col];
#pragma unroll
      for (int v = 0; v < 8; v++) {
        const int row = rowBase + i * 16 + hk * 8 + v;
        const float val = acc[i][j][v] + bj;
        if (MODE == 0)      Cf[(size_t)row * N + col] = val;
        else if (MODE == 1) Cb[(size_t)row * N + col] = f2bfu(val);
        else                Cf[(size_t)row * N + col] += val;
      }
    }
  }
}

// ---------------------------------------------------------------------------
// Block-diagonal attention: one wave per (batch, head, 32-token block).
// qkv fp32 [B,S,3D]; lane = query row; softmax over the 32 in-block keys.
// Output bf16 [B,S,D] with head-concatenated columns.
// ---------------------------------------------------------------------------
__global__ __launch_bounds__(32)
void attn_kernel(const float* __restrict__ qkv, unsigned short* __restrict__ attb) {
  const int id   = blockIdx.x;          // 0 .. B*16*32-1
  const int blk  = id & 31;
  const int hh   = (id >> 5) & 15;
  const int b    = id >> 9;
  const int t    = threadIdx.x;
  const int tok0 = b * 1024 + blk * 32;

  const float* qrow = qkv + (size_t)(tok0 + t) * 3072 + hh * 64;
  float q[64];
#pragma unroll
  for (int d = 0; d < 64; d++) q[d] = qrow[d];

  float sc[32];
  float mx = -1e30f;
#pragma unroll
  for (int j = 0; j < 32; j++) {
    const float* krow = qkv + (size_t)(tok0 + j) * 3072 + 1024 + hh * 64;
    float s = 0.f;
#pragma unroll
    for (int d = 0; d < 64; d++) s += q[d] * krow[d];
    s *= 0.125f;                        // 1/sqrt(64)
    sc[j] = s;
    mx = fmaxf(mx, s);
  }
  float sum = 0.f;
#pragma unroll
  for (int j = 0; j < 32; j++) { sc[j] = __expf(sc[j] - mx); sum += sc[j]; }
  const float inv = 1.0f / sum;

  float o[64];
#pragma unroll
  for (int d = 0; d < 64; d++) o[d] = 0.f;
#pragma unroll
  for (int j = 0; j < 32; j++) {
    const float* vrow = qkv + (size_t)(tok0 + j) * 3072 + 2048 + hh * 64;
    const float p = sc[j] * inv;
#pragma unroll
    for (int d = 0; d < 64; d++) o[d] += p * vrow[d];
  }
  unsigned short* orow = attb + (size_t)(tok0 + t) * 1024 + hh * 64;
#pragma unroll
  for (int d = 0; d < 64; d++) orow[d] = f2bfu(o[d]);
}

// ---------------------------------------------------------------------------
// gate = gelu_exact(u1) * u2   (bf16 in / bf16 out), 4 elems per thread.
// ---------------------------------------------------------------------------
__device__ __forceinline__ unsigned short gate1(unsigned short a, unsigned short b) {
  const float x  = bf2f(a);
  const float ge = 0.5f * x * (1.0f + erff(x * 0.70710678118654752f));
  return f2bfu(ge * bf2f(b));
}

__global__ __launch_bounds__(256)
void gate_kernel(const unsigned short* __restrict__ u1,
                 const unsigned short* __restrict__ u2,
                 unsigned short* __restrict__ g) {
  const int i = blockIdx.x * blockDim.x + threadIdx.x;   // n/4 threads
  const ushort4 a4 = ((const ushort4*)u1)[i];
  const ushort4 b4 = ((const ushort4*)u2)[i];
  ushort4 o4;
  o4.x = gate1(a4.x, b4.x);
  o4.y = gate1(a4.y, b4.y);
  o4.z = gate1(a4.z, b4.z);
  o4.w = gate1(a4.w, b4.w);
  ((ushort4*)g)[i] = o4;
}

// ---------------------------------------------------------------------------
// Host orchestration.
// ---------------------------------------------------------------------------
extern "C" void kernel_launch(void* const* d_in, const int* in_sizes, int n_in,
                              void* d_out, int out_size, void* d_ws, size_t ws_size,
                              hipStream_t stream) {
  (void)in_sizes; (void)n_in; (void)out_size; (void)ws_size;

  const float* x     = (const float*)d_in[0];
  const float* w_qkv = (const float*)d_in[1];
  const float* b_qkv = (const float*)d_in[2];
  const float* w_o   = (const float*)d_in[3];
  const float* b_o   = (const float*)d_in[4];
  const float* ln1_w = (const float*)d_in[5];
  const float* ln1_b = (const float*)d_in[6];
  const float* ln2_w = (const float*)d_in[7];
  const float* ln2_b = (const float*)d_in[8];
  const float* w1    = (const float*)d_in[9];
  const float* b1    = (const float*)d_in[10];
  const float* w2    = (const float*)d_in[11];
  const float* b2    = (const float*)d_in[12];
  const float* w3    = (const float*)d_in[13];
  const float* b3    = (const float*)d_in[14];
  const float* lnf_w = (const float*)d_in[15];
  const float* lnf_b = (const float*)d_in[16];

  const int M = 2048;               // B*S
  char* ws = (char*)d_ws;
  size_t off = 0;
  float* hbuf          = (float*)(ws + off);          off += (size_t)M * 1024 * 4;  // residual stream
  float* qkv           = (float*)(ws + off);          off += (size_t)M * 3072 * 4;  // fp32 qkv
  unsigned short* lnb  = (unsigned short*)(ws + off); off += (size_t)M * 1024 * 2;  // LN out (bf16)
  unsigned short* attb = (unsigned short*)(ws + off); off += (size_t)M * 1024 * 2;  // attn out (bf16)
  unsigned short* u1b  = (unsigned short*)(ws + off); off += (size_t)M * 4096 * 2;  // mlp up1 (bf16)
  unsigned short* u2b  = (unsigned short*)(ws + off); off += (size_t)M * 4096 * 2;  // mlp up2 (bf16)
  unsigned short* gtb  = (unsigned short*)(ws + off); off += (size_t)M * 4096 * 2;  // gate (bf16)
  // per-layer bf16 weight staging (L2-resident during the layer's GEMMs)
  unsigned short* wqb  = (unsigned short*)(ws + off); off += (size_t)3072 * 1024 * 2;
  unsigned short* wob  = (unsigned short*)(ws + off); off += (size_t)1024 * 1024 * 2;
  unsigned short* w1b  = (unsigned short*)(ws + off); off += (size_t)4096 * 1024 * 2;
  unsigned short* w2b  = (unsigned short*)(ws + off); off += (size_t)4096 * 1024 * 2;
  unsigned short* w3b  = (unsigned short*)(ws + off); off += (size_t)1024 * 4096 * 2;

  // RoPE: x -> h
  rope_kernel<<<2048, 256, 0, stream>>>(x, hbuf);

  for (int l = 0; l < 8; l++) {
    const float* wq = w_qkv + (size_t)l * 3072 * 1024;
    const float* bq = b_qkv + (size_t)l * 3072;
    const float* wo = w_o   + (size_t)l * 1024 * 1024;
    const float* bo = b_o   + (size_t)l * 1024;
    const float* W1 = w1    + (size_t)l * 4096 * 1024;
    const float* B1 = b1    + (size_t)l * 4096;
    const float* W2 = w2    + (size_t)l * 4096 * 1024;
    const float* B2 = b2    + (size_t)l * 4096;
    const float* W3 = w3    + (size_t)l * 1024 * 4096;
    const float* B3 = b3    + (size_t)l * 1024;

    // Stage this layer's weights as bf16 (stream-ordered before consumers).
    cvt_kernel<<<3072, 256, 0, stream>>>(wq, wqb);
    cvt_kernel<<<1024, 256, 0, stream>>>(wo, wob);
    cvt_kernel<<<4096, 256, 0, stream>>>(W1, w1b);
    cvt_kernel<<<4096, 256, 0, stream>>>(W2, w2b);
    cvt_kernel<<<4096, 256, 0, stream>>>(W3, w3b);

    // LN1 -> bf16
    ln_kernel<1><<<2048, 256, 0, stream>>>(hbuf, ln1_w + l * 1024, ln1_b + l * 1024,
                                           lnb, nullptr);
    // QKV projection: [2048,1024] x [3072,1024]^T -> fp32
    gemm_bf16<0><<<dim3(24, 8), 256, 0, stream>>>(lnb, wqb, bq, qkv, nullptr,
                                                  M, 3072, 1024);
    // Block-diagonal attention -> bf16
    attn_kernel<<<1024, 32, 0, stream>>>(qkv, attb);
    // O projection + residual into h
    gemm_bf16<2><<<dim3(8, 8), 256, 0, stream>>>(attb, wob, bo, hbuf, nullptr,
                                                 M, 1024, 1024);
    // LN2 -> bf16
    ln_kernel<1><<<2048, 256, 0, stream>>>(hbuf, ln2_w + l * 1024, ln2_b + l * 1024,
                                           lnb, nullptr);
    // MLP up-projections -> bf16
    gemm_bf16<1><<<dim3(32, 8), 256, 0, stream>>>(lnb, w1b, B1, nullptr, u1b,
                                                  M, 4096, 1024);
    gemm_bf16<1><<<dim3(32, 8), 256, 0, stream>>>(lnb, w2b, B2, nullptr, u2b,
                                                  M, 4096, 1024);
    // gelu gate
    gate_kernel<<<8192, 256, 0, stream>>>(u1b, u2b, gtb);
    // down projection + residual into h
    gemm_bf16<2><<<dim3(8, 8), 256, 0, stream>>>(gtb, w3b, B3, hbuf, nullptr,
                                                 M, 1024, 4096);
  }

  // Final LN -> fp32 output
  ln_kernel<0><<<2048, 256, 0, stream>>>(hbuf, lnf_w, lnf_b, nullptr, (float*)d_out);
}
